// AttentionModified_146028888652
// MI455X (gfx1250) — matrix operand order repaired and checked by reading, not compiled, not run on hardware
//
#include <hip/hip_runtime.h>

// ---------------------------------------------------------------------------
// AttentionModified for MI455X (gfx1250, wave32, WMMA)
//   pass 0: convert x + all weights f32 -> bf16 (one touch each)
//   pass 1: q  = x @ Wq^T        (WMMA bf16, B staged in LDS, async dbl-buf)
//   pass 2: kv = vp @ Wkv^T      (fused (V,B)->(B,V) remap; variants read once)
//   pass 3: tiny softmax-attention (VALU, ~0.2 GFLOP)
//   pass 4: out = ao @ Wproj^T + b
// ---------------------------------------------------------------------------

typedef __attribute__((ext_vector_type(16))) __bf16 v16bf;
typedef __attribute__((ext_vector_type(8)))  __bf16 v8bf;
typedef __attribute__((ext_vector_type(8)))  float  v8f;

// Types spelled to match the async-LDS builtin's expected parameters:
// arg0: 'int __attribute__((vector_size(16))) __device__ *'  (global, AS1)
// arg1: LDS side (AS3)
typedef int v4i __attribute__((vector_size(16)));
typedef __attribute__((address_space(1))) v4i* as1_v4i_p;
typedef __attribute__((address_space(3))) v4i* as3_v4i_p;

#define DEV_ static __device__ __forceinline__

constexpr int V_   = 8;
constexpr int B_   = 8;
constexpr int NSEQ = 1024;
constexpr int C_   = 768;
constexpr int H_   = 12;
constexpr int HD_  = 64;
constexpr int KVW  = 2 * HD_;        // 128
constexpr int MQ   = B_ * NSEQ;      // 8192
constexpr int MKV  = B_ * V_ * NSEQ; // 65536

// LDS B-tile row pitch: 32 K-elems + 8 pad -> 20 dwords; 20*n mod 64 distinct
// for n in [0,16) -> conflict-free ds_load_b128 per lane.
constexpr int LP = 40;

// ---- async global->LDS (guarded; sync ds_store fallback) -------------------
#if defined(__has_builtin)
#if __has_builtin(__builtin_amdgcn_global_load_async_to_lds_b128)
#define HAS_ASYNC_LDS 1
#endif
#endif
#ifndef HAS_ASYNC_LDS
#define HAS_ASYNC_LDS 0
#endif

DEV_ void cp16_g2l(const __bf16* __restrict__ g, __bf16* l) {
#if HAS_ASYNC_LDS
  // Global flat address == AS1 address on amdgcn; LDS flat address truncates
  // to the low 32 bits for the AS3 offset (ISA 10.2). Integer round-trips
  // sidestep address-space cast restrictions.
  as1_v4i_p gp = (as1_v4i_p)(unsigned long long)g;
  as3_v4i_p lp = (as3_v4i_p)(unsigned int)(unsigned long long)l;
  __builtin_amdgcn_global_load_async_to_lds_b128(gp, lp, 0, 0);
#else
  *(v8bf*)l = *(const v8bf*)g;
#endif
}

DEV_ void stage_fence() {
#if HAS_ASYNC_LDS
#if __has_builtin(__builtin_amdgcn_s_wait_asynccnt)
  __builtin_amdgcn_s_wait_asynccnt(0);
#else
  asm volatile("s_wait_asynccnt 0x0" ::: "memory");
#endif
#endif
  __syncthreads();
}

// Stage `rows` x 32K bf16 tile (from row-major panel, ld=C_) into LDS.
// rows*4 chunks of 16B, strided across 256 threads.
DEV_ void stage_tile(const __bf16* __restrict__ panel, __bf16* lds, int k0,
                     int tid, int rows) {
  for (int c = tid; c < rows * 4; c += 256) {
    const int r = c >> 2, col = c & 3;
    cp16_g2l(panel + (size_t)r * C_ + k0 + col * 8, lds + r * LP + col * 8);
  }
}

// ---- WMMA fragment loaders -------------------------------------------------
// A (16x32 bf16): lane%16 = M; lanes 0-15 hold K {0..7,16..23},
//                 lanes 16-31 hold K {8..15,24..31}.
DEV_ v16bf frag_a_bf16(const __bf16* __restrict__ rowA, int k0, int hi) {
  const __bf16* p = rowA + k0 + (hi ? 8 : 0);
  v8bf c0 = *(const v8bf*)(p);
  v8bf c1 = *(const v8bf*)(p + 16);
  v16bf a;
#pragma unroll
  for (int j = 0; j < 8; ++j) { a[j] = c0[j]; a[j + 8] = c1[j]; }
  return a;
}

DEV_ v16bf frag_a_f32(const float* __restrict__ rowA, int k0, int hi) {
  const float* p = rowA + k0 + (hi ? 8 : 0);
  float buf[16];
  *(float4*)(buf + 0)  = *(const float4*)(p + 0);
  *(float4*)(buf + 4)  = *(const float4*)(p + 4);
  *(float4*)(buf + 8)  = *(const float4*)(p + 16);
  *(float4*)(buf + 12) = *(const float4*)(p + 20);
  v16bf a;
#pragma unroll
  for (int j = 0; j < 16; ++j) a[j] = (__bf16)buf[j];
  return a;
}

// B (32x16 K x N) from LDS tile: lane%16 = N col; lanes 0-15 K {0..7,16..23},
// lanes 16-31 K {8..15,24..31} (2x ds_load_b128 per lane).
DEV_ v16bf frag_b_lds(const __bf16* lds, int nLocal, int hi) {
  const __bf16* p = lds + nLocal * LP + (hi ? 8 : 0);
  v8bf c0 = *(const v8bf*)(p);
  v8bf c1 = *(const v8bf*)(p + 16);
  v16bf b;
#pragma unroll
  for (int j = 0; j < 8; ++j) { b[j] = c0[j]; b[j + 8] = c1[j]; }
  return b;
}

// ---- pass 0: f32 -> bf16 convert ------------------------------------------
__global__ void __launch_bounds__(256)
k_cvt_bf16(const float* __restrict__ src, __bf16* __restrict__ dst, int n) {
  const int i = (blockIdx.x * 256 + threadIdx.x) * 8;
  if (i >= n) return;
  float4 f0 = *(const float4*)(src + i);
  float4 f1 = *(const float4*)(src + i + 4);
  v8bf o;
  o[0] = (__bf16)f0.x; o[1] = (__bf16)f0.y; o[2] = (__bf16)f0.z; o[3] = (__bf16)f0.w;
  o[4] = (__bf16)f1.x; o[5] = (__bf16)f1.y; o[6] = (__bf16)f1.z; o[7] = (__bf16)f1.w;
  *(v8bf*)(dst + i) = o;
}

// ---- pass 1: q = x @ Wq^T (A bf16, B via async LDS, out bf16) --------------
__global__ void __launch_bounds__(256)
k_gemm_q(const __bf16* __restrict__ Xb, const __bf16* __restrict__ Wb,
         __bf16* __restrict__ Q) {
  __shared__ __bf16 ldsB[2][64 * LP];
  const int tid  = threadIdx.x;
  const int wave = tid >> 5, lane = tid & 31;
  const int hi   = (lane >> 4) & 1;
  const int m0   = blockIdx.x * 128 + wave * 16;
  const int n0   = blockIdx.y * 64;

  const __bf16* rowA   = Xb + (size_t)(m0 + (lane & 15)) * C_;
  const __bf16* panelB = Wb + (size_t)n0 * C_;

  stage_tile(panelB, ldsB[0], 0, tid, 64);
  stage_fence();

  v8f acc[4];
#pragma unroll
  for (int t = 0; t < 4; ++t) acc[t] = {};

  int cur = 0;
  for (int k0 = 0; k0 < C_; k0 += 32) {
    if (k0 + 32 < C_) stage_tile(panelB, ldsB[cur ^ 1], k0 + 32, tid, 64);
    v16bf a = frag_a_bf16(rowA, k0, hi);
#pragma unroll
    for (int t = 0; t < 4; ++t) {
      v16bf b = frag_b_lds(ldsB[cur], t * 16 + (lane & 15), hi);
      acc[t] = __builtin_amdgcn_wmma_f32_16x16x32_bf16(
          false, a, false, b, (short)0, acc[t], false, false);
    }
    stage_fence();
    cur ^= 1;
  }
  const int nc = n0 + (lane & 15);
  const int mr = m0 + (hi ? 8 : 0);
#pragma unroll
  for (int t = 0; t < 4; ++t)
#pragma unroll
    for (int r = 0; r < 8; ++r)
      Q[(size_t)(mr + r) * C_ + nc + t * 16] = (__bf16)acc[t][r];
}

// ---- pass 2: kv = vp @ Wkv^T (A f32 read-once + remap, out bf16) -----------
__global__ void __launch_bounds__(256)
k_gemm_kv(const float* __restrict__ VP, const __bf16* __restrict__ Wb,
          __bf16* __restrict__ KV) {
  __shared__ __bf16 ldsB[2][128 * LP];
  const int tid  = threadIdx.x;
  const int wave = tid >> 5, lane = tid & 31;
  const int hi   = (lane >> 4) & 1;
  const int m0   = blockIdx.x * 128 + wave * 16;  // logical (b, v, n) row
  const int b    = m0 / (V_ * NSEQ);
  const int vv   = (m0 / NSEQ) % V_;
  const int nn   = m0 % NSEQ;                     // 16-row tile: same (b,v)

  const float* rowA =
      VP + ((size_t)(vv * B_ + b) * NSEQ + nn + (lane & 15)) * C_;

  stage_tile(Wb, ldsB[0], 0, tid, 128);
  stage_fence();

  v8f acc[8];
#pragma unroll
  for (int t = 0; t < 8; ++t) acc[t] = {};

  int cur = 0;
  for (int k0 = 0; k0 < C_; k0 += 32) {
    if (k0 + 32 < C_) {
      __builtin_prefetch(rowA + k0 + 32, 0, 1);
      stage_tile(Wb, ldsB[cur ^ 1], k0 + 32, tid, 128);
    }
    v16bf a = frag_a_f32(rowA, k0, hi);
#pragma unroll
    for (int t = 0; t < 8; ++t) {
      v16bf bfr = frag_b_lds(ldsB[cur], t * 16 + (lane & 15), hi);
      acc[t] = __builtin_amdgcn_wmma_f32_16x16x32_bf16(
          false, a, false, bfr, (short)0, acc[t], false, false);
    }
    stage_fence();
    cur ^= 1;
  }
  const int nc = (lane & 15);
  const int mr = m0 + (hi ? 8 : 0);
#pragma unroll
  for (int t = 0; t < 8; ++t)
#pragma unroll
    for (int r = 0; r < 8; ++r)
      KV[(size_t)(mr + r) * KVW + nc + t * 16] = (__bf16)acc[t][r];
}

// ---- pass 3: softmax attention core (one thread per (b,n,h)) ---------------
__global__ void __launch_bounds__(256)
k_attn(const __bf16* __restrict__ Q, const __bf16* __restrict__ KV,
       __bf16* __restrict__ AO) {
  const int gid = blockIdx.x * 256 + threadIdx.x;
  if (gid >= B_ * NSEQ * H_) return;
  const int h = gid % H_;
  const int n = (gid / H_) % NSEQ;
  const int b = gid / (H_ * NSEQ);

  const __bf16* qrow = Q + ((size_t)(b * NSEQ + n) * C_) + h * HD_;
  float qr[HD_];
#pragma unroll
  for (int j = 0; j < HD_; j += 8) {
    v8bf t = *(const v8bf*)(qrow + j);
#pragma unroll
    for (int e = 0; e < 8; ++e) qr[j + e] = (float)t[e];
  }

  const float scale = 0.125f;  // 64^-0.5
  float s[V_];
  float mx = -3.0e38f;
#pragma unroll
  for (int vv = 0; vv < V_; ++vv) {
    const __bf16* krow = KV + (size_t)((b * V_ + vv) * NSEQ + n) * KVW;
    float d = 0.f;
#pragma unroll
    for (int j = 0; j < HD_; j += 8) {
      v8bf t = *(const v8bf*)(krow + j);
#pragma unroll
      for (int e = 0; e < 8; ++e) d += qr[j + e] * (float)t[e];
    }
    s[vv] = d * scale;
    mx = fmaxf(mx, s[vv]);
  }
  float sum = 0.f;
#pragma unroll
  for (int vv = 0; vv < V_; ++vv) { s[vv] = __expf(s[vv] - mx); sum += s[vv]; }
  const float inv = 1.f / sum;

  float o[HD_];
#pragma unroll
  for (int j = 0; j < HD_; ++j) o[j] = 0.f;
#pragma unroll
  for (int vv = 0; vv < V_; ++vv) {
    const float p = s[vv] * inv;
    const __bf16* vrow = KV + (size_t)((b * V_ + vv) * NSEQ + n) * KVW + HD_;
#pragma unroll
    for (int j = 0; j < HD_; j += 8) {
      v8bf t = *(const v8bf*)(vrow + j);
#pragma unroll
      for (int e = 0; e < 8; ++e) o[j + e] += p * (float)t[e];
    }
  }
  __bf16* orow = AO + ((size_t)(b * NSEQ + n) * C_) + h * HD_;
#pragma unroll
  for (int j = 0; j < HD_; j += 8) {
    v8bf t;
#pragma unroll
    for (int e = 0; e < 8; ++e) t[e] = (__bf16)o[j + e];
    *(v8bf*)(orow + j) = t;
  }
}

// ---- pass 4: out = ao @ Wproj^T + bproj (out f32) --------------------------
__global__ void __launch_bounds__(256)
k_gemm_proj(const __bf16* __restrict__ AO, const __bf16* __restrict__ Wb,
            const float* __restrict__ bias, float* __restrict__ OUT) {
  __shared__ __bf16 ldsB[2][64 * LP];
  const int tid  = threadIdx.x;
  const int wave = tid >> 5, lane = tid & 31;
  const int hi   = (lane >> 4) & 1;
  const int m0   = blockIdx.x * 128 + wave * 16;
  const int n0   = blockIdx.y * 64;

  const __bf16* rowA   = AO + (size_t)(m0 + (lane & 15)) * C_;
  const __bf16* panelB = Wb + (size_t)n0 * C_;

  stage_tile(panelB, ldsB[0], 0, tid, 64);
  stage_fence();

  v8f acc[4];
#pragma unroll
  for (int t = 0; t < 4; ++t) acc[t] = {};

  int cur = 0;
  for (int k0 = 0; k0 < C_; k0 += 32) {
    if (k0 + 32 < C_) stage_tile(panelB, ldsB[cur ^ 1], k0 + 32, tid, 64);
    v16bf a = frag_a_bf16(rowA, k0, hi);
#pragma unroll
    for (int t = 0; t < 4; ++t) {
      v16bf b = frag_b_lds(ldsB[cur], t * 16 + (lane & 15), hi);
      acc[t] = __builtin_amdgcn_wmma_f32_16x16x32_bf16(
          false, a, false, b, (short)0, acc[t], false, false);
    }
    stage_fence();
    cur ^= 1;
  }
  const int nc = n0 + (lane & 15);
  const int mr = m0 + (hi ? 8 : 0);
#pragma unroll
  for (int t = 0; t < 4; ++t) {
    const float bv = bias[nc + t * 16];
#pragma unroll
    for (int r = 0; r < 8; ++r)
      OUT[(size_t)(mr + r) * C_ + nc + t * 16] = acc[t][r] + bv;
  }
}

// ---------------------------------------------------------------------------
extern "C" void kernel_launch(void* const* d_in, const int* in_sizes, int n_in,
                              void* d_out, int out_size, void* d_ws,
                              size_t ws_size, hipStream_t stream) {
  const float* x     = (const float*)d_in[0];
  const float* vp    = (const float*)d_in[1];
  const float* Wq    = (const float*)d_in[2];
  const float* Wkv   = (const float*)d_in[3];
  const float* Wproj = (const float*)d_in[4];
  const float* bproj = (const float*)d_in[5];
  float* out = (float*)d_out;

  // Workspace layout (all sizes 256B-aligned multiples):
  char* ws = (char*)d_ws;
  const size_t xN  = (size_t)MQ * C_;        // 6,291,456
  const size_t kvN = (size_t)MKV * KVW;      // 8,388,608
  const size_t wqN = (size_t)C_ * C_;        // 589,824
  const size_t wkN = (size_t)KVW * C_;       // 98,304
  __bf16* xbf  = (__bf16*)(ws);
  __bf16* qbf  = xbf + xN;
  __bf16* kvbf = qbf + xN;
  __bf16* aobf = kvbf + kvN;
  __bf16* wqbf = aobf + xN;
  __bf16* wkbf = wqbf + wqN;
  __bf16* wpbf = wkbf + wkN;

  dim3 blk(256);
  // pass 0: conversions (one touch each)
  k_cvt_bf16<<<dim3((unsigned)(xN / 8 / 256)), blk, 0, stream>>>(x, xbf, (int)xN);
  k_cvt_bf16<<<dim3((unsigned)(wqN / 8 / 256)), blk, 0, stream>>>(Wq, wqbf, (int)wqN);
  k_cvt_bf16<<<dim3((unsigned)(wkN / 8 / 256)), blk, 0, stream>>>(Wkv, wkbf, (int)wkN);
  k_cvt_bf16<<<dim3((unsigned)(wqN / 8 / 256)), blk, 0, stream>>>(Wproj, wpbf, (int)wqN);
  // passes 1..4
  k_gemm_q<<<dim3(MQ / 128, C_ / 64), blk, 0, stream>>>(xbf, wqbf, qbf);
  k_gemm_kv<<<dim3(MKV / 128), blk, 0, stream>>>(vp, wkbf, kvbf);
  const int attnThreads = B_ * NSEQ * H_;
  k_attn<<<dim3((attnThreads + 255) / 256), blk, 0, stream>>>(qbf, kvbf, aobf);
  k_gemm_proj<<<dim3(MQ / 128, C_ / 64), blk, 0, stream>>>(aobf, wpbf, bproj,
                                                           out);
}
